// NodeLevelActorCritic_36721970381076
// MI455X (gfx1250) — matrix operand order repaired and checked by reading, not compiled, not run on hardware
//
#include <hip/hip_runtime.h>
#include <hip/hip_bf16.h>

typedef __bf16 bf16;
typedef __attribute__((ext_vector_type(16))) __bf16 v16bf;
typedef __attribute__((ext_vector_type(8)))  float  v8f;

union FragA { v16bf v; bf16 e[16]; };

#define N_T   10
#define N_R   16
#define DD    128
#define HH    256
#define BB    16384
#define E_T   20
#define E_R   48
#define FBd   6
#define GDIM  268      /* 2*D + 6 + FB */
#define NPAIR 45
#define K_PF  288      /* 257 padded to multiple of 32 */
#define NLOG  56       /* 1 noop + 10 split + 45 merge */

__constant__ int c_PI[NPAIR] = {0,0,0,0,0,0,0,0,0,1,1,1,1,1,1,1,1,2,2,2,2,2,2,2,
                                3,3,3,3,3,3,4,4,4,4,4,5,5,5,5,6,6,6,7,7,8};
__constant__ int c_PJ[NPAIR] = {1,2,3,4,5,6,7,8,9,2,3,4,5,6,7,8,9,3,4,5,6,7,8,9,
                                4,5,6,7,8,9,5,6,7,8,9,6,7,8,9,7,8,9,8,9,9};

__device__ __forceinline__ float block_sum(float v, float* red) {
    int t = threadIdx.x;
    red[t] = v; __syncthreads();
    for (int s = 128; s > 0; s >>= 1) { if (t < s) red[t] += red[t + s]; __syncthreads(); }
    float r = red[0]; __syncthreads();
    return r;
}

// ---------------------------------------------------------------------------
// Generic bf16 WMMA GEMM:  Y = A@W^T (+ A2@W2^T) + bias (+Cadd) (*rowscale) (ReLU)
// A: MxK row-major bf16, W: NxK row-major bf16.
// One wave computes a 16x64 strip (4 accumulators) so each A fragment is
// reused by 4 (DUAL: 8) consecutive v_wmma issues per 32-wide K-step.
// ---------------------------------------------------------------------------
template<int DUAL>
__global__ __launch_bounds__(128) void gemm_wmma(
    const bf16* __restrict__ A,  const bf16* __restrict__ W,
    const bf16* __restrict__ A2, const bf16* __restrict__ W2,
    const float* __restrict__ bias, const float* __restrict__ Cadd,
    const float* __restrict__ rowscale,
    float* __restrict__ Yf, bf16* __restrict__ Yh,
    int M, int N, int K, int relu)
{
    int lane = threadIdx.x & 31;
    int wave = blockIdx.x * 4 + (threadIdx.x >> 5);
    int Nstr = N >> 6;                     // 64-wide strips
    int strips = (M >> 4) * Nstr;
    if (wave >= strips) return;
    int mt = wave / Nstr, ns = wave % Nstr;
    int l15 = lane & 15, hi = lane >> 4;
    const int pb[8] = {0,2,4,6,8,10,12,14};   // A-fragment pair K-offsets (contiguous layout)

    v8f c[4];
#pragma unroll
    for (int n = 0; n < 4; ++n)
#pragma unroll
        for (int r = 0; r < 8; ++r) c[n][r] = 0.f;

    const bf16* arow  = A + (size_t)(mt*16 + l15) * K;
    const bf16* arow2 = DUAL ? (A2 + (size_t)(mt*16 + l15) * K) : (const bf16*)0;
    const bf16* wrow0 = W + (size_t)(ns*64 + l15) * K + (hi << 4);
    const bf16* wrow2 = DUAL ? (W2 + (size_t)(ns*64 + l15) * K + (hi << 4)) : (const bf16*)0;
    const size_t wstep = (size_t)16 * K;   // next 16 output columns

    for (int kt = 0; kt < K; kt += 32) {
        FragA a;
#pragma unroll
        for (int r = 0; r < 8; ++r) {
            int kk = kt + pb[r] + (hi << 4);
            a.e[2*r]   = arow[kk];
            a.e[2*r+1] = arow[kk+1];
        }
        FragA a2;
        if (DUAL) {
#pragma unroll
            for (int r = 0; r < 8; ++r) {
                int kk = kt + pb[r] + (hi << 4);
                a2.e[2*r]   = arow2[kk];
                a2.e[2*r+1] = arow2[kk+1];
            }
        }
#pragma unroll
        for (int n = 0; n < 4; ++n) {
            FragA bfr;
#pragma unroll
            for (int e = 0; e < 16; ++e) bfr.e[e] = wrow0[n*wstep + kt + e];
            c[n] = __builtin_amdgcn_wmma_f32_16x16x32_bf16(false, a.v, false, bfr.v,
                                                           (short)0, c[n], false, false);
            if (DUAL) {
                FragA b2;
#pragma unroll
                for (int e = 0; e < 16; ++e) b2.e[e] = wrow2[n*wstep + kt + e];
                c[n] = __builtin_amdgcn_wmma_f32_16x16x32_bf16(false, a2.v, false, b2.v,
                                                               (short)0, c[n], false, false);
            }
        }
    }

#pragma unroll
    for (int n = 0; n < 4; ++n) {
        int col = ns*64 + n*16 + l15;
        float bv = bias ? bias[col] : 0.f;
#pragma unroll
        for (int r = 0; r < 8; ++r) {
            int row = mt*16 + r + (hi << 3);
            float v = c[n][r] + bv;
            if (Cadd)     v += Cadd[(size_t)row * N + col];
            if (rowscale) v *= rowscale[row];
            if (relu)     v = fmaxf(v, 0.f);
            if (Yf) Yf[(size_t)row * N + col] = v;
            if (Yh) Yh[(size_t)row * N + col] = (bf16)v;
        }
    }
}

// ---------------------------------------------------------------------------
// GNN layer 0 (K = 2 or 3): segment-mean + two tiny matvecs + layernorm + relu.
// One block per batch element.
// ---------------------------------------------------------------------------
__global__ __launch_bounds__(256) void gnn_layer0(
    const float* __restrict__ X, const int* __restrict__ E,
    const float* __restrict__ Wl, const float* __restrict__ Wr,
    const float* __restrict__ linb, const float* __restrict__ g, const float* __restrict__ bsh,
    bf16* __restrict__ xfOut, int NN, int FF, int EE, int b0)
{
    __shared__ float xs[N_R*3];
    __shared__ float agg[N_R*3];
    __shared__ int   eds[2*E_R];
    __shared__ int   cnt[N_R];
    __shared__ float hbuf[N_R*HH];
    __shared__ float red[256];
    int t = threadIdx.x, b = blockIdx.x;
    const float* xb = X + (size_t)(b0 + b) * NN * FF;
    const int*   eb = E + (size_t)(b0 + b) * 2 * EE;
    for (int i = t; i < NN*FF; i += 256) { xs[i] = xb[i]; agg[i] = 0.f; }
    for (int i = t; i < 2*EE;  i += 256) eds[i] = eb[i];
    if (t < NN) cnt[t] = 0;
    __syncthreads();
    if (t == 0) {
        for (int e = 0; e < EE; ++e) {
            int s = eds[e], d = eds[EE + e];
            cnt[d]++;
            for (int f = 0; f < FF; ++f) agg[d*FF + f] += xs[s*FF + f];
        }
    }
    __syncthreads();
    for (int i = t; i < NN*HH; i += 256) {
        int n = i >> 8, o = i & 255;
        float den = fmaxf((float)cnt[n], 1.f);
        float acc = linb[o];
        for (int f = 0; f < FF; ++f)
            acc += (agg[n*FF + f] / den) * Wl[o*FF + f] + xs[n*FF + f] * Wr[o*FF + f];
        hbuf[i] = acc;
    }
    __syncthreads();
    for (int n = 0; n < NN; ++n) {
        float v   = hbuf[n*HH + t];
        float m   = block_sum(v, red) * (1.f/HH);
        float d   = v - m;
        float var = block_sum(d*d, red) * (1.f/HH);
        float y   = d * rsqrtf(var + 1e-5f) * g[t] + bsh[t];
        xfOut[(size_t)(b*NN + n)*HH + t] = (bf16)fmaxf(y, 0.f);
    }
}

// Segment-mean of layer-0 activations (input to layer-1 WMMA GEMM).
__global__ __launch_bounds__(256) void gnn_agg(
    const bf16* __restrict__ xf, const int* __restrict__ E,
    bf16* __restrict__ an, int NN, int EE, int b0)
{
    __shared__ int eds[2*E_R];
    __shared__ int cnt[N_R];
    int t = threadIdx.x, b = blockIdx.x;
    const int* eb = E + (size_t)(b0 + b) * 2 * EE;
    for (int i = t; i < 2*EE; i += 256) eds[i] = eb[i];
    if (t < NN) cnt[t] = 0;
    __syncthreads();
    if (t == 0) { for (int e = 0; e < EE; ++e) cnt[eds[EE + e]]++; }
    __syncthreads();
    for (int i = t; i < NN*HH; i += 256) {
        int n = i >> 8, k = i & 255;
        float acc = 0.f;
        for (int e = 0; e < EE; ++e)
            if (eds[EE + e] == n) acc += (float)xf[(size_t)(b*NN + eds[e])*HH + k];
        an[(size_t)(b*NN + n)*HH + k] = (bf16)(acc / fmaxf((float)cnt[n], 1.f));
    }
}

// Row layernorm (width 256) + relu -> bf16
__global__ __launch_bounds__(256) void ln_relu(
    const float* __restrict__ Hb, const float* __restrict__ g, const float* __restrict__ bsh,
    bf16* __restrict__ out)
{
    __shared__ float red[256];
    int t = threadIdx.x; size_t row = blockIdx.x;
    float v   = Hb[row*HH + t];
    float m   = block_sum(v, red) * (1.f/HH);
    float d   = v - m;
    float var = block_sum(d*d, red) * (1.f/HH);
    float y   = d * rsqrtf(var + 1e-5f) * g[t] + bsh[t];
    out[row*HH + t] = (bf16)fmaxf(y, 0.f);
}

__global__ void mask_kernel(const float* __restrict__ tx, float* __restrict__ real, int C, int b0) {
    int i = blockIdx.x*256 + threadIdx.x;
    if (i < C*N_T) {
        size_t gi = (size_t)(b0*N_T) + i;
        float a = fabsf(tx[gi*2]) + fabsf(tx[gi*2 + 1]);
        real[i] = (a > 0.f) ? 1.f : 0.f;
    }
}

// ---------------------------------------------------------------------------
// Attention: per-batch wave. scores = Q·K^T/sqrt(D) via 4 WMMA k-steps,
// LDS softmax, ctx = attn·V via one zero-padded WMMA k-step per N-tile.
// ---------------------------------------------------------------------------
__global__ __launch_bounds__(128) void attn_kernel(
    const bf16* __restrict__ Qb, const bf16* __restrict__ Kb, const bf16* __restrict__ Vb,
    float* __restrict__ attnG, bf16* __restrict__ ctxB)
{
    __shared__ float s[4][16][16];
    int ws = threadIdx.x >> 5, lane = threadIdx.x & 31;
    int b  = blockIdx.x * 4 + ws;
    int l15 = lane & 15, hi = lane >> 4;
    const int pb[8] = {0,2,4,6,8,10,12,14};

    v8f c;
#pragma unroll
    for (int r = 0; r < 8; ++r) c[r] = 0.f;
    const bf16* qrow = Qb + (size_t)(b*N_T + l15) * DD;   // rows >= 10 read pad/next-batch: discarded
    const bf16* krow = Kb + (size_t)(b*N_R + l15) * DD;
    for (int kt = 0; kt < DD; kt += 32) {
        FragA a, bb_;
#pragma unroll
        for (int r = 0; r < 8; ++r) {
            int kk = kt + pb[r] + (hi << 4);
            a.e[2*r] = qrow[kk]; a.e[2*r+1] = qrow[kk+1];
        }
        int ks = kt + (hi << 4);
#pragma unroll
        for (int e = 0; e < 16; ++e) bb_.e[e] = krow[ks + e];
        c = __builtin_amdgcn_wmma_f32_16x16x32_bf16(false, a.v, false, bb_.v, (short)0, c, false, false);
    }
    const float scale = 0.0883883476483184f; // 1/sqrt(128)
#pragma unroll
    for (int r = 0; r < 8; ++r) s[ws][r + (hi << 3)][l15] = c[r] * scale;
    __syncthreads();
    if (lane < N_T) {
        float mx = -1e30f;
        for (int j = 0; j < N_R; ++j) mx = fmaxf(mx, s[ws][lane][j]);
        float e[N_R], sum = 0.f;
        for (int j = 0; j < N_R; ++j) { e[j] = __expf(s[ws][lane][j] - mx); sum += e[j]; }
        float inv = 1.f / sum;
        for (int j = 0; j < N_R; ++j) {
            float av = e[j] * inv;
            s[ws][lane][j] = av;
            attnG[(size_t)(b*N_T + lane)*N_R + j] = av;
        }
    }
    __syncthreads();
    FragA a2;
#pragma unroll
    for (int r = 0; r < 8; ++r) {
        int k0 = pb[r] + (hi << 4);
        a2.e[2*r]   = (k0     < N_R) ? (bf16)s[ws][l15][k0]     : (bf16)0.f;
        a2.e[2*r+1] = (k0 + 1 < N_R) ? (bf16)s[ws][l15][k0 + 1] : (bf16)0.f;
    }
    for (int nt = 0; nt < DD/16; ++nt) {
        FragA bb_;
#pragma unroll
        for (int e = 0; e < 16; ++e) {
            int k = (hi << 4) + e;
            bb_.e[e] = (k < N_R) ? Vb[(size_t)(b*N_R + k)*DD + nt*16 + l15] : (bf16)0.f;
        }
        v8f c2;
#pragma unroll
        for (int r = 0; r < 8; ++r) c2[r] = 0.f;
        c2 = __builtin_amdgcn_wmma_f32_16x16x32_bf16(false, a2.v, false, bb_.v, (short)0, c2, false, false);
#pragma unroll
        for (int r = 0; r < 8; ++r) {
            int row = r + (hi << 3);
            if (row < N_T) ctxB[(size_t)(b*N_T + row)*DD + nt*16 + l15] = (bf16)c2[r];
        }
    }
}

// pf = [emb_i(128) | emb_j(128) | lq | zero-pad...] per pair, K padded to 288
__global__ __launch_bounds__(288) void build_pf(
    const bf16* __restrict__ enrB, const float* __restrict__ attnG, bf16* __restrict__ pf)
{
    __shared__ float slq;
    int bp = blockIdx.x, t = threadIdx.x;
    int b = bp / NPAIR, p = bp % NPAIR;
    int i = c_PI[p], j = c_PJ[p];
    if (t == 0) {
        const float* ai = attnG + (size_t)(b*N_T + i)*N_R;
        const float* aj = attnG + (size_t)(b*N_T + j)*N_R;
        float acc = 0.f;
        for (int r = 0; r < N_R; ++r) acc += ai[r]*aj[r];
        slq = acc;
    }
    __syncthreads();
    bf16 v;
    if      (t < DD)      v = enrB[(size_t)(b*N_T + i)*DD + t];
    else if (t < 2*DD)    v = enrB[(size_t)(b*N_T + j)*DD + (t - DD)];
    else if (t == 2*DD)   v = (bf16)slq;
    else                  v = (bf16)0.f;
    pf[(size_t)bp*K_PF + t] = v;
}

__global__ void dot_m2(const bf16* __restrict__ hmid, const float* __restrict__ w,
                       const float* __restrict__ bia, float* __restrict__ out, int C, int b0) {
    int i = blockIdx.x*256 + threadIdx.x;
    if (i >= C*NPAIR) return;
    int b = i / NPAIR, p = i % NPAIR;
    float acc = 0.f;
    for (int k = 0; k < DD; ++k) acc += (float)hmid[(size_t)i*DD + k] * w[k];
    out[(size_t)(b0 + b)*NLOG + 11 + p] = acc + bia[0];
}

__global__ void dot_split(const float* __restrict__ enrF, const float* __restrict__ w,
                          const float* __restrict__ bia, float* __restrict__ out, int C, int b0) {
    int i = blockIdx.x*256 + threadIdx.x;
    if (i >= C*N_T) return;
    int b = i / N_T, n = i % N_T;
    float acc = 0.f;
    for (int k = 0; k < DD; ++k) acc += enrF[(size_t)i*DD + k] * w[k];
    out[(size_t)(b0 + b)*NLOG + 1 + n] = acc + bia[0];
}

__global__ __launch_bounds__(256) void global_head(
    const float* __restrict__ enrF, const float* __restrict__ rnF,
    const float* __restrict__ rx, const float* __restrict__ tx, const float* __restrict__ fb,
    const float* __restrict__ noop_w, const float* __restrict__ noop_b,
    const float* __restrict__ val_w,  const float* __restrict__ val_b,
    float* __restrict__ out, int b0)
{
    __shared__ float gs[GDIM];
    __shared__ float red[256];
    __shared__ float nreal;
    int t = threadIdx.x, b = blockIdx.x, gb = b0 + b;
    if (t == 0) {
        float cc = 0.f;
        for (int n = 0; n < N_T; ++n) {
            size_t gi = (size_t)(gb*N_T + n);
            float a = fabsf(tx[gi*2]) + fabsf(tx[gi*2 + 1]);
            cc += (a > 0.f) ? 1.f : 0.f;
        }
        nreal = fmaxf(cc, 1.f);
    }
    __syncthreads();
    if (t < DD) {
        float s = 0.f;
        for (int n = 0; n < N_T; ++n) s += enrF[(size_t)(b*N_T + n)*DD + t];
        gs[t] = s / nreal;
    } else {
        int d = t - DD;
        float s = 0.f;
        for (int n = 0; n < N_R; ++n) s += rnF[(size_t)(b*N_R + n)*DD + d];
        gs[DD + d] = s * (1.f/N_R);
    }
    if (t < 3) {
        float mn = 1e30f, sm = 0.f;
        for (int n = 0; n < N_R; ++n) {
            float v = rx[(size_t)(gb*N_R + n)*3 + t];
            mn = fminf(mn, v); sm += v;
        }
        gs[2*DD + t]     = mn;
        gs[2*DD + 3 + t] = sm * (1.f/N_R);
    }
    if (t < FBd) gs[2*DD + 6 + t] = fb[(size_t)gb*FBd + t];
    __syncthreads();
    float pn = (t < GDIM ? gs[t]*noop_w[t] : 0.f) + (t + 256 < GDIM ? gs[t+256]*noop_w[t+256] : 0.f);
    float sn = block_sum(pn, red);
    float pv = (t < GDIM ? gs[t]*val_w[t]  : 0.f) + (t + 256 < GDIM ? gs[t+256]*val_w[t+256]  : 0.f);
    float sv = block_sum(pv, red);
    if (t == 0) {
        out[(size_t)gb*NLOG]      = sn + noop_b[0];
        out[(size_t)BB*NLOG + gb] = sv + val_b[0];
    }
}

__global__ void f2bf(const float* __restrict__ s, bf16* __restrict__ d, int n) {
    int i = blockIdx.x*256 + threadIdx.x;
    if (i < n) d[i] = (bf16)s[i];
}
__global__ void f2bf_pad(const float* __restrict__ s, bf16* __restrict__ d, int rows, int kin, int kout) {
    int i = blockIdx.x*256 + threadIdx.x;
    if (i < rows*kout) {
        int r = i / kout, k = i % kout;
        d[i] = (k < kin) ? (bf16)s[(size_t)r*kin + k] : (bf16)0.f;
    }
}

// ---------------------------------------------------------------------------
// Host launcher. Input order = JAX pytree flatten (sorted dict keys).
// ---------------------------------------------------------------------------
extern "C" void kernel_launch(void* const* d_in, const int* in_sizes, int n_in,
                              void* d_out, int out_size, void* d_ws, size_t ws_size,
                              hipStream_t stream)
{
    (void)in_sizes; (void)n_in; (void)out_size;
    const float* task_x    = (const float*)d_in[0];
    const int*   task_edge = (const int*)  d_in[1];
    const float* res_x     = (const float*)d_in[2];
    const int*   res_edge  = (const int*)  d_in[3];
    const float* feedback  = (const float*)d_in[4];
    const float* p_k    = (const float*)d_in[5];
    const float* p_m1b  = (const float*)d_in[6];
    const float* p_m1w  = (const float*)d_in[7];
    const float* p_m2b  = (const float*)d_in[8];
    const float* p_m2w  = (const float*)d_in[9];
    const float* p_noopb= (const float*)d_in[10];
    const float* p_noopw= (const float*)d_in[11];
    const float* p_ob   = (const float*)d_in[12];
    const float* p_ow   = (const float*)d_in[13];
    const float* p_q    = (const float*)d_in[14];
    const float* rWl0=(const float*)d_in[15], *rWr0=(const float*)d_in[16];
    const float* r_b0=(const float*)d_in[17], *r_bl0=(const float*)d_in[18], *r_g0=(const float*)d_in[19];
    const float* rWl1=(const float*)d_in[20], *rWr1=(const float*)d_in[21];
    const float* r_b1=(const float*)d_in[22], *r_bl1=(const float*)d_in[23], *r_g1=(const float*)d_in[24];
    const float* r_pjb=(const float*)d_in[25], *r_pjw=(const float*)d_in[26];
    const float* p_splitb=(const float*)d_in[27], *p_splitw=(const float*)d_in[28];
    const float* tWl0=(const float*)d_in[29], *tWr0=(const float*)d_in[30];
    const float* t_b0=(const float*)d_in[31], *t_bl0=(const float*)d_in[32], *t_g0=(const float*)d_in[33];
    const float* tWl1=(const float*)d_in[34], *tWr1=(const float*)d_in[35];
    const float* t_b1=(const float*)d_in[36], *t_bl1=(const float*)d_in[37], *t_g1=(const float*)d_in[38];
    const float* t_pjb=(const float*)d_in[39], *t_pjw=(const float*)d_in[40];
    const float* p_v   = (const float*)d_in[41];
    const float* p_valb= (const float*)d_in[42], *p_valw=(const float*)d_in[43];
    float* out = (float*)d_out;

    char* base = (char*)d_ws;
    size_t off = 0;
    auto alloc = [&](size_t bytes)->void* {
        void* p = base + off;
        off = (off + bytes + 8192 + 255) & ~(size_t)255;   // +slack rows for padded frag reads
        return p;
    };
    // persistent bf16 weights
    bf16* wtWl1 = (bf16*)alloc(65536*2);
    bf16* wtWr1 = (bf16*)alloc(65536*2);
    bf16* wrWl1 = (bf16*)alloc(65536*2);
    bf16* wrWr1 = (bf16*)alloc(65536*2);
    bf16* wtPj  = (bf16*)alloc(32768*2);
    bf16* wrPj  = (bf16*)alloc(32768*2);
    bf16* wQ    = (bf16*)alloc(16384*2);
    bf16* wK    = (bf16*)alloc(16384*2);
    bf16* wV    = (bf16*)alloc(16384*2);
    bf16* wO    = (bf16*)alloc(16384*2);
    bf16* wM1p  = (bf16*)alloc(128*K_PF*2);
    size_t off_persist = off;

    auto chunkBytes = [&](long C)->size_t {
        size_t s = 0;
        auto add = [&](size_t b){ s += (b + 8192 + 255) & ~(size_t)255; };
        add((size_t)C*2560*2);  add((size_t)C*2560*2);              // xfT anT
        add((size_t)C*4096*2);  add((size_t)C*4096*2);              // xfR anR
        add((size_t)C*2560*4);  add((size_t)C*4096*4);              // hT hR
        add((size_t)C*1280*4);  add((size_t)C*1280*2);              // tnF tnB
        add((size_t)C*2048*4);  add((size_t)C*2048*2);              // rnF rnB
        add((size_t)C*1280*2);  add((size_t)C*2048*2);  add((size_t)C*2048*2); // Qb Kb Vb
        add((size_t)C*160*4);   add((size_t)C*1280*2);              // attn ctxB
        add((size_t)C*1280*4);  add((size_t)C*1280*2);              // enrF enrB
        add((size_t)C*NPAIR*K_PF*2); add((size_t)C*NPAIR*DD*2);     // pf hmid
        add((size_t)C*N_T*4);                                       // real
        return s;
    };
    long C = 16;
    const long cands[7] = {2048,1024,512,256,128,64,32};
    for (int i = 0; i < 7; ++i)
        if (off_persist + chunkBytes(cands[i]) <= ws_size) { C = cands[i]; break; }

    bf16* xfT  = (bf16*)alloc((size_t)C*2560*2);
    bf16* anT  = (bf16*)alloc((size_t)C*2560*2);
    bf16* xfR  = (bf16*)alloc((size_t)C*4096*2);
    bf16* anR  = (bf16*)alloc((size_t)C*4096*2);
    float* hT  = (float*)alloc((size_t)C*2560*4);
    float* hR  = (float*)alloc((size_t)C*4096*4);
    float* tnF = (float*)alloc((size_t)C*1280*4);
    bf16*  tnB = (bf16*) alloc((size_t)C*1280*2);
    float* rnF = (float*)alloc((size_t)C*2048*4);
    bf16*  rnB = (bf16*) alloc((size_t)C*2048*2);
    bf16*  Qb  = (bf16*) alloc((size_t)C*1280*2);
    bf16*  Kb  = (bf16*) alloc((size_t)C*2048*2);
    bf16*  Vb  = (bf16*) alloc((size_t)C*2048*2);
    float* attnG=(float*)alloc((size_t)C*160*4);
    bf16*  ctxB =(bf16*) alloc((size_t)C*1280*2);
    float* enrF =(float*)alloc((size_t)C*1280*4);
    bf16*  enrB =(bf16*) alloc((size_t)C*1280*2);
    bf16*  pf   =(bf16*) alloc((size_t)C*NPAIR*K_PF*2);
    bf16*  hmid =(bf16*) alloc((size_t)C*NPAIR*DD*2);
    float* realb=(float*)alloc((size_t)C*N_T*4);

    // weight conversion to bf16 (deterministic: runs every call)
    f2bf<<<(65536+255)/256,256,0,stream>>>(tWl1, wtWl1, 65536);
    f2bf<<<(65536+255)/256,256,0,stream>>>(tWr1, wtWr1, 65536);
    f2bf<<<(65536+255)/256,256,0,stream>>>(rWl1, wrWl1, 65536);
    f2bf<<<(65536+255)/256,256,0,stream>>>(rWr1, wrWr1, 65536);
    f2bf<<<(32768+255)/256,256,0,stream>>>(t_pjw, wtPj, 32768);
    f2bf<<<(32768+255)/256,256,0,stream>>>(r_pjw, wrPj, 32768);
    f2bf<<<(16384+255)/256,256,0,stream>>>(p_q, wQ, 16384);
    f2bf<<<(16384+255)/256,256,0,stream>>>(p_k, wK, 16384);
    f2bf<<<(16384+255)/256,256,0,stream>>>(p_v, wV, 16384);
    f2bf<<<(16384+255)/256,256,0,stream>>>(p_ow, wO, 16384);
    f2bf_pad<<<(128*K_PF+255)/256,256,0,stream>>>(p_m1w, wM1p, 128, 257, K_PF);

    for (int b0 = 0; b0 < BB; b0 += (int)C) {
        int Ci = (int)C;
        // --- task GNN ---
        gnn_layer0<<<Ci,256,0,stream>>>(task_x, task_edge, tWl0, tWr0, t_bl0, t_g0, t_b0, xfT, N_T, 2, E_T, b0);
        gnn_agg<<<Ci,256,0,stream>>>(xfT, task_edge, anT, N_T, E_T, b0);
        { int M = Ci*N_T, st = (M/16)*(HH/64);
          gemm_wmma<1><<<(st+3)/4,128,0,stream>>>(anT, wtWl1, xfT, wtWr1, t_bl1, nullptr, nullptr, hT, nullptr, M, HH, HH, 0); }
        ln_relu<<<Ci*N_T,256,0,stream>>>(hT, t_g1, t_b1, xfT);
        mask_kernel<<<(Ci*N_T+255)/256,256,0,stream>>>(task_x, realb, Ci, b0);
        { int M = Ci*N_T, st = (M/16)*(DD/64);
          gemm_wmma<0><<<(st+3)/4,128,0,stream>>>(xfT, wtPj, nullptr, nullptr, t_pjb, nullptr, realb, tnF, tnB, M, DD, HH, 0); }
        // --- res GNN ---
        gnn_layer0<<<Ci,256,0,stream>>>(res_x, res_edge, rWl0, rWr0, r_bl0, r_g0, r_b0, xfR, N_R, 3, E_R, b0);
        gnn_agg<<<Ci,256,0,stream>>>(xfR, res_edge, anR, N_R, E_R, b0);
        { int M = Ci*N_R, st = (M/16)*(HH/64);
          gemm_wmma<1><<<(st+3)/4,128,0,stream>>>(anR, wrWl1, xfR, wrWr1, r_bl1, nullptr, nullptr, hR, nullptr, M, HH, HH, 0); }
        ln_relu<<<Ci*N_R,256,0,stream>>>(hR, r_g1, r_b1, xfR);
        { int M = Ci*N_R, st = (M/16)*(DD/64);
          gemm_wmma<0><<<(st+3)/4,128,0,stream>>>(xfR, wrPj, nullptr, nullptr, r_pjb, nullptr, nullptr, rnF, rnB, M, DD, HH, 0); }
        // --- attention ---
        { int M = Ci*N_T, st = (M/16)*(DD/64);
          gemm_wmma<0><<<(st+3)/4,128,0,stream>>>(tnB, wQ, nullptr, nullptr, nullptr, nullptr, nullptr, nullptr, Qb, M, DD, DD, 0); }
        { int M = Ci*N_R, st = (M/16)*(DD/64);
          gemm_wmma<0><<<(st+3)/4,128,0,stream>>>(rnB, wK, nullptr, nullptr, nullptr, nullptr, nullptr, nullptr, Kb, M, DD, DD, 0); }
        { int M = Ci*N_R, st = (M/16)*(DD/64);
          gemm_wmma<0><<<(st+3)/4,128,0,stream>>>(rnB, wV, nullptr, nullptr, nullptr, nullptr, nullptr, nullptr, Vb, M, DD, DD, 0); }
        attn_kernel<<<Ci/4,128,0,stream>>>(Qb, Kb, Vb, attnG, ctxB);
        { int M = Ci*N_T, st = (M/16)*(DD/64);   // enriched = task_node + ctx@o_w^T + o_b
          gemm_wmma<0><<<(st+3)/4,128,0,stream>>>(ctxB, wO, nullptr, nullptr, p_ob, tnF, nullptr, enrF, enrB, M, DD, DD, 0); }
        // --- heads ---
        build_pf<<<Ci*NPAIR,K_PF,0,stream>>>(enrB, attnG, pf);
        { int M = Ci*NPAIR, st = (M/16)*(DD/64);
          gemm_wmma<0><<<(st+3)/4,128,0,stream>>>(pf, wM1p, nullptr, nullptr, p_m1b, nullptr, nullptr, nullptr, hmid, M, DD, K_PF, 1); }
        dot_m2<<<(Ci*NPAIR+255)/256,256,0,stream>>>(hmid, p_m2w, p_m2b, out, Ci, b0);
        dot_split<<<(Ci*N_T+255)/256,256,0,stream>>>(enrF, p_splitw, p_splitb, out, Ci, b0);
        global_head<<<Ci,256,0,stream>>>(enrF, rnF, res_x, task_x, feedback,
                                         p_noopw, p_noopb, p_valw, p_valb, out, b0);
    }
}
